// NEGAN_45423574123155
// MI455X (gfx1250) — compile-verified, matrix operand order checked by reading, not run on hardware
//
#include <hip/hip_runtime.h>
#include <hip/hip_bf16.h>
#include <math.h>

// ---------------------------------------------------------------------------
// CDNA5 / gfx1250 bf16 WMMA types and helpers
// ---------------------------------------------------------------------------
typedef __attribute__((ext_vector_type(16))) __bf16          v16bf;
typedef __attribute__((ext_vector_type(16))) unsigned short  v16us;
typedef __attribute__((ext_vector_type(8)))  float           v8f;
typedef __attribute__((ext_vector_type(8)))  unsigned int    v8u;
typedef __attribute__((ext_vector_type(2)))  float           f2v;

__device__ __forceinline__ unsigned short f2bf(float f) {
    // round-to-nearest-even f32 -> bf16
    unsigned int u = __builtin_bit_cast(unsigned int, f);
    u += 0x7FFFu + ((u >> 16) & 1u);
    return (unsigned short)(u >> 16);
}
__device__ __forceinline__ unsigned int pack2(float lo, float hi) {
    return (unsigned int)f2bf(lo) | ((unsigned int)f2bf(hi) << 16);
}

// ---------------------------------------------------------------------------
// Tiled GEMM:  C[M,N] = act(A[M,K] * B[N,K]^T + bias[N])
// A, B are f32 row-major; converted to bf16 in LDS; v_wmma_f32_16x16x32_bf16.
// Block = 256 threads (8 waves), tile 128(M) x 64(N) x 32(K);
// each wave computes a 32x32 block (2x2 WMMA tiles).
// Staging: block-uniform fast path (unguarded float2 loads) for interior
// tiles; clamp+select (branch-free) on clipped tiles.
// ---------------------------------------------------------------------------
#define GBM 128
#define GBN 64
#define GBK 32

__global__ __launch_bounds__(256)
void gemm_bias_relu(const float* __restrict__ A, int lda,
                    const float* __restrict__ B, int ldb,
                    const float* __restrict__ bias,
                    float* __restrict__ C, int ldc,
                    int M, int N, int K, int doRelu)
{
    __shared__ unsigned int As[GBM][GBK / 2];   // bf16 pairs, 8 KB
    __shared__ unsigned int Bs[GBN][GBK / 2];   // bf16 pairs, 4 KB

    const int tid  = threadIdx.x;
    const int wave = tid >> 5;
    const int lane = tid & 31;
    const int wm   = wave >> 1;          // 0..3 (M dir, 32 rows each)
    const int wn   = wave & 1;           // 0..1 (N dir, 32 cols each)
    const int bm0  = blockIdx.x * GBM;
    const int bn0  = blockIdx.y * GBN;
    const int r    = lane & 15;
    const int kh   = lane >> 4;

    v8f acc[2][2] = {};

    for (int k0 = 0; k0 < K; k0 += GBK) {
        const bool fullK = (k0 + GBK <= K);
        // ---- stage A tile ----
        if (fullK && (bm0 + GBM <= M)) {
            for (int p = tid; p < GBM * (GBK / 2); p += 256) {
                int row = p >> 4, kk = (p & 15) << 1;
                f2v f = *(const f2v*)(A + (size_t)(bm0 + row) * lda + (k0 + kk));
                As[row][p & 15] = pack2(f.x, f.y);
            }
        } else {
            for (int p = tid; p < GBM * (GBK / 2); p += 256) {
                int row = p >> 4, kk = (p & 15) << 1;
                int gm = bm0 + row, gk = k0 + kk;
                int cm = gm < M ? gm : M - 1;
                int c0 = gk < K ? gk : K - 1;
                int c1 = gk + 1 < K ? gk + 1 : K - 1;
                float f0 = A[(size_t)cm * lda + c0];
                float f1 = A[(size_t)cm * lda + c1];
                if (!(gm < M && gk     < K)) f0 = 0.f;
                if (!(gm < M && gk + 1 < K)) f1 = 0.f;
                As[row][p & 15] = pack2(f0, f1);
            }
        }
        // ---- stage B tile ----
        if (fullK && (bn0 + GBN <= N)) {
            for (int p = tid; p < GBN * (GBK / 2); p += 256) {
                int row = p >> 4, kk = (p & 15) << 1;
                f2v f = *(const f2v*)(B + (size_t)(bn0 + row) * ldb + (k0 + kk));
                Bs[row][p & 15] = pack2(f.x, f.y);
            }
        } else {
            for (int p = tid; p < GBN * (GBK / 2); p += 256) {
                int row = p >> 4, kk = (p & 15) << 1;
                int gn = bn0 + row, gk = k0 + kk;
                int cn = gn < N ? gn : N - 1;
                int c0 = gk < K ? gk : K - 1;
                int c1 = gk + 1 < K ? gk + 1 : K - 1;
                float f0 = B[(size_t)cn * ldb + c0];
                float f1 = B[(size_t)cn * ldb + c1];
                if (!(gn < N && gk     < K)) f0 = 0.f;
                if (!(gn < N && gk + 1 < K)) f1 = 0.f;
                Bs[row][p & 15] = pack2(f0, f1);
            }
        }
        __syncthreads();

        // A/B fragments: ISA 16-bit 16x32 layout.
        // element idx -> k = ((idx>>3)<<4) + kh*8 + (idx&7); pair dword idx:
        // ((v>>2)<<3) + kh*4 + (v&3)
        v16bf afrag[2], bfrag[2];
        #pragma unroll
        for (int im = 0; im < 2; ++im) {
            int mrow = wm * 32 + im * 16 + r;
            v8u au;
            #pragma unroll
            for (int v = 0; v < 8; ++v)
                au[v] = As[mrow][((v >> 2) << 3) + kh * 4 + (v & 3)];
            afrag[im] = __builtin_bit_cast(v16bf, au);
        }
        #pragma unroll
        for (int in2 = 0; in2 < 2; ++in2) {
            int nrow = wn * 32 + in2 * 16 + r;
            v8u bu;
            #pragma unroll
            for (int v = 0; v < 8; ++v)
                bu[v] = Bs[nrow][((v >> 2) << 3) + kh * 4 + (v & 3)];
            bfrag[in2] = __builtin_bit_cast(v16bf, bu);
        }
        #pragma unroll
        for (int im = 0; im < 2; ++im)
            #pragma unroll
            for (int in2 = 0; in2 < 2; ++in2)
                acc[im][in2] = __builtin_amdgcn_wmma_f32_16x16x32_bf16(
                    false, afrag[im], false, bfrag[in2],
                    (short)0, acc[im][in2], false, false);
        __syncthreads();
    }

    // C/D layout: lanes 0-15: M=v, N=lane; lanes 16-31: M=8+v, N=lane-16
    const int nn = lane & 15;
    const int mb = (lane >> 4) << 3;
    #pragma unroll
    for (int im = 0; im < 2; ++im)
        for (int in2 = 0; in2 < 2; ++in2)
            #pragma unroll
            for (int v = 0; v < 8; ++v) {
                int gm = bm0 + wm * 32 + im * 16 + mb + v;
                int gn = bn0 + wn * 32 + in2 * 16 + nn;
                if (gm < M && gn < N) {
                    float x = acc[im][in2][v] + (bias ? bias[gn] : 0.f);
                    if (doRelu) x = fmaxf(x, 0.f);
                    C[(size_t)gm * ldc + gn] = x;
                }
            }
}

// ---------------------------------------------------------------------------
// Strided-batched WMMA GEMM (116^3 per sample) with transpose modes, LDS-staged.
//   aKM=0: Aop[m,k]=a[m*lda+k]   aKM=1: Aop[m,k]=a[k*lda+m]
//   bKN=0: Bop[k,n]=b[n*ldb+k]   bKN=1: Bop[k,n]=b[k*ldb+n]
// Block = 128 threads (4 waves), tile 64(M) x 64(N) x 32(K);
// each wave computes a 32x32 block (2x2 WMMA tiles). Tails zero-padded in LDS.
// ---------------------------------------------------------------------------
#define BBM 64
#define BBN 64
#define BBK 32

__global__ __launch_bounds__(128)
void bgemm116(const float* __restrict__ A, int lda, long long sA, int aKM,
              const float* __restrict__ B, int ldb, long long sB, int bKN,
              const float* __restrict__ biasM,
              float* __restrict__ C, int ldc, long long sC,
              int M, int N, int K, int nb)
{
    __shared__ unsigned int As[BBM][BBK / 2];   // 4 KB
    __shared__ unsigned int Bs[BBN][BBK / 2];   // 4 KB

    const int tM  = (M + BBM - 1) / BBM;
    const int tN  = (N + BBN - 1) / BBN;
    const int per = tM * tN;
    const int batch = blockIdx.x / per;
    if (batch >= nb) return;                    // whole block returns uniformly
    const int t  = blockIdx.x - batch * per;
    const int m0 = (t / tN) * BBM;
    const int n0 = (t % tN) * BBN;

    const float* a = A + (size_t)batch * sA;
    const float* b = B + (size_t)batch * sB;

    const int tid  = threadIdx.x;
    const int wave = tid >> 5;
    const int lane = tid & 31;
    const int wm   = wave >> 1;          // 0..1
    const int wn   = wave & 1;           // 0..1
    const int r    = lane & 15;
    const int kh   = lane >> 4;

    v8f acc[2][2] = {};

    for (int k0 = 0; k0 < K; k0 += BBK) {
        const bool fullK = (k0 + BBK <= K);
        const bool fullA = fullK && (m0 + BBM <= M);
        const bool fullB = fullK && (n0 + BBN <= N);
        // ---- stage A tile (mode-aware) ----
        for (int p = tid; p < BBM * (BBK / 2); p += 128) {
            int row = p >> 4, kk = (p & 15) << 1;
            int gm = m0 + row, gk = k0 + kk;
            float f0, f1;
            if (fullA) {
                if (aKM) {
                    f0 = a[(size_t)gk * lda + gm];
                    f1 = a[(size_t)(gk + 1) * lda + gm];
                } else {
                    f2v f = *(const f2v*)(a + (size_t)gm * lda + gk);
                    f0 = f.x; f1 = f.y;
                }
            } else {
                int cm = gm < M ? gm : M - 1;
                int c0 = gk < K ? gk : K - 1;
                int c1 = gk + 1 < K ? gk + 1 : K - 1;
                if (aKM) {
                    f0 = a[(size_t)c0 * lda + cm];
                    f1 = a[(size_t)c1 * lda + cm];
                } else {
                    f0 = a[(size_t)cm * lda + c0];
                    f1 = a[(size_t)cm * lda + c1];
                }
                if (!(gm < M && gk     < K)) f0 = 0.f;
                if (!(gm < M && gk + 1 < K)) f1 = 0.f;
            }
            As[row][p & 15] = pack2(f0, f1);
        }
        // ---- stage B tile (mode-aware) ----
        for (int p = tid; p < BBN * (BBK / 2); p += 128) {
            int row = p >> 4, kk = (p & 15) << 1;
            int gn = n0 + row, gk = k0 + kk;
            float f0, f1;
            if (fullB) {
                if (bKN) {
                    f0 = b[(size_t)gk * ldb + gn];
                    f1 = b[(size_t)(gk + 1) * ldb + gn];
                } else {
                    f2v f = *(const f2v*)(b + (size_t)gn * ldb + gk);
                    f0 = f.x; f1 = f.y;
                }
            } else {
                int cn = gn < N ? gn : N - 1;
                int c0 = gk < K ? gk : K - 1;
                int c1 = gk + 1 < K ? gk + 1 : K - 1;
                if (bKN) {
                    f0 = b[(size_t)c0 * ldb + cn];
                    f1 = b[(size_t)c1 * ldb + cn];
                } else {
                    f0 = b[(size_t)cn * ldb + c0];
                    f1 = b[(size_t)cn * ldb + c1];
                }
                if (!(gn < N && gk     < K)) f0 = 0.f;
                if (!(gn < N && gk + 1 < K)) f1 = 0.f;
            }
            Bs[row][p & 15] = pack2(f0, f1);
        }
        __syncthreads();

        v16bf afrag[2], bfrag[2];
        #pragma unroll
        for (int im = 0; im < 2; ++im) {
            int mrow = wm * 32 + im * 16 + r;
            v8u au;
            #pragma unroll
            for (int v = 0; v < 8; ++v)
                au[v] = As[mrow][((v >> 2) << 3) + kh * 4 + (v & 3)];
            afrag[im] = __builtin_bit_cast(v16bf, au);
        }
        #pragma unroll
        for (int in2 = 0; in2 < 2; ++in2) {
            int nrow = wn * 32 + in2 * 16 + r;
            v8u bu;
            #pragma unroll
            for (int v = 0; v < 8; ++v)
                bu[v] = Bs[nrow][((v >> 2) << 3) + kh * 4 + (v & 3)];
            bfrag[in2] = __builtin_bit_cast(v16bf, bu);
        }
        #pragma unroll
        for (int im = 0; im < 2; ++im)
            #pragma unroll
            for (int in2 = 0; in2 < 2; ++in2)
                acc[im][in2] = __builtin_amdgcn_wmma_f32_16x16x32_bf16(
                    false, afrag[im], false, bfrag[in2],
                    (short)0, acc[im][in2], false, false);
        __syncthreads();
    }

    const int nn = lane & 15;
    const int mb = (lane >> 4) << 3;
    float* c = C + (size_t)batch * sC;
    #pragma unroll
    for (int im = 0; im < 2; ++im)
        for (int in2 = 0; in2 < 2; ++in2)
            #pragma unroll
            for (int v = 0; v < 8; ++v) {
                int gm = m0 + wm * 32 + im * 16 + mb + v;
                int gn = n0 + wn * 32 + in2 * 16 + nn;
                if (gm < M && gn < N) {
                    float x = acc[im][in2][v];
                    if (biasM) x += biasM[gm];
                    c[(size_t)gm * ldc + gn] = x;
                }
            }
}

// ---------------------------------------------------------------------------
// Attention: K = conv1_w @ X[s]^T + b (3x116); att = softmax(K^T K) rows.
// One block per sample.
// ---------------------------------------------------------------------------
__global__ __launch_bounds__(128)
void attn_kernel(const float* __restrict__ X, const float* __restrict__ w,
                 const float* __restrict__ bias, float* __restrict__ att)
{
    int s = blockIdx.x;
    __shared__ float Ks[3][120];
    const float* x = X + (size_t)s * 348;
    float w00 = w[0], w01 = w[1], w02 = w[2];
    float w10 = w[3], w11 = w[4], w12 = w[5];
    float w20 = w[6], w21 = w[7], w22 = w[8];
    float b0 = bias[0], b1 = bias[1], b2 = bias[2];
    for (int l = threadIdx.x; l < 116; l += blockDim.x) {
        float x0 = x[l * 3], x1 = x[l * 3 + 1], x2 = x[l * 3 + 2];
        Ks[0][l] = w00 * x0 + w01 * x1 + w02 * x2 + b0;
        Ks[1][l] = w10 * x0 + w11 * x1 + w12 * x2 + b1;
        Ks[2][l] = w20 * x0 + w21 * x1 + w22 * x2 + b2;
    }
    __syncthreads();
    for (int l = threadIdx.x; l < 116; l += blockDim.x) {
        float k0 = Ks[0][l], k1 = Ks[1][l], k2 = Ks[2][l];
        float mx = -3.4e38f;
        for (int m = 0; m < 116; ++m) {
            float sv = k0 * Ks[0][m] + k1 * Ks[1][m] + k2 * Ks[2][m];
            mx = fmaxf(mx, sv);
        }
        float* ar = att + (size_t)s * 13456 + (size_t)l * 116;
        float sum = 0.f;
        for (int m = 0; m < 116; ++m) {
            float sv = k0 * Ks[0][m] + k1 * Ks[1][m] + k2 * Ks[2][m];
            float e = __expf(sv - mx);
            sum += e;
            ar[m] = e;
        }
        float inv = 1.f / sum;
        for (int m = 0; m < 116; ++m) ar[m] *= inv;
    }
}

// ---------------------------------------------------------------------------
// BatchNorm over [n,C,L] (axes 0,2): per-channel scale/shift. 1 block/channel.
// ---------------------------------------------------------------------------
__global__ __launch_bounds__(256)
void bn_ncl_stats(const float* __restrict__ Y, int n, int C, int Ldim,
                  const float* __restrict__ g, const float* __restrict__ b,
                  float* __restrict__ scale, float* __restrict__ shift)
{
    int c = blockIdx.x;
    __shared__ float rs[256], rq[256];
    float sum = 0.f, sq = 0.f;
    long long tot = (long long)n * Ldim;
    for (long long idx = threadIdx.x; idx < tot; idx += blockDim.x) {
        int s = (int)(idx / Ldim), l = (int)(idx % Ldim);
        float v = Y[(size_t)s * C * Ldim + (size_t)c * Ldim + l];
        sum += v; sq += v * v;
    }
    rs[threadIdx.x] = sum; rq[threadIdx.x] = sq;
    __syncthreads();
    for (int st = 128; st > 0; st >>= 1) {
        if ((int)threadIdx.x < st) {
            rs[threadIdx.x] += rs[threadIdx.x + st];
            rq[threadIdx.x] += rq[threadIdx.x + st];
        }
        __syncthreads();
    }
    if (threadIdx.x == 0) {
        float inv = 1.f / (float)tot;
        float m = rs[0] * inv;
        float var = rq[0] * inv - m * m;
        float rr = rsqrtf(var + 1e-5f);
        float sc = g[c] * rr;
        scale[c] = sc;
        shift[c] = b[c] - m * sc;
    }
}

// IO += relu(Y*scale[c] + shift[c])   over [n,C,L]
__global__ void bn_resid_relu_add(const float* __restrict__ Y, float* __restrict__ IO,
                                  const float* __restrict__ scale,
                                  const float* __restrict__ shift,
                                  int C, int Ldim, long long total)
{
    long long i = (long long)blockIdx.x * blockDim.x + threadIdx.x;
    if (i >= total) return;
    int rem = (int)(i % (long long)(C * Ldim));
    int c = rem / Ldim;
    float v = Y[i] * scale[c] + shift[c];
    IO[i] += fmaxf(v, 0.f);
}

// ---------------------------------------------------------------------------
// Node branch: X1[s,i,:] = (sum_j P[s,j,i] * X[s,j,:]) @ node_w  (3-wide)
// ---------------------------------------------------------------------------
__global__ void node_kernel(const float* __restrict__ P, const float* __restrict__ X,
                            const float* __restrict__ nw, float* __restrict__ X1, int n)
{
    int gid = blockIdx.x * blockDim.x + threadIdx.x;
    int s = gid / 116, i = gid % 116;
    if (s >= n) return;
    const float* p = P + (size_t)s * 13456;
    const float* x = X + (size_t)s * 348;
    float u0 = 0.f, u1 = 0.f, u2 = 0.f;
    for (int j = 0; j < 116; ++j) {
        float pj = p[(size_t)j * 116 + i];
        u0 = fmaf(pj, x[j * 3],     u0);
        u1 = fmaf(pj, x[j * 3 + 1], u1);
        u2 = fmaf(pj, x[j * 3 + 2], u2);
    }
    float* o = X1 + (size_t)s * 348 + i * 3;
    #pragma unroll
    for (int l = 0; l < 3; ++l)
        o[l] = u0 * nw[l] + u1 * nw[3 + l] + u2 * nw[6 + l];
}

// ---------------------------------------------------------------------------
// Column-wise BatchNorm over [rows, cols] (bn_nc), g/b split ln/le. 1 blk/col.
// ---------------------------------------------------------------------------
__global__ __launch_bounds__(256)
void bn_cols(float* __restrict__ XZ, int rows, int cols,
             const float* __restrict__ gln, const float* __restrict__ bln,
             const float* __restrict__ gle, const float* __restrict__ ble, int split)
{
    int j = blockIdx.x;
    __shared__ float rs[256], rq[256];
    __shared__ float scs, shs;
    float sum = 0.f, sq = 0.f;
    for (int r2 = threadIdx.x; r2 < rows; r2 += blockDim.x) {
        float v = XZ[(size_t)r2 * cols + j];
        sum += v; sq += v * v;
    }
    rs[threadIdx.x] = sum; rq[threadIdx.x] = sq;
    __syncthreads();
    for (int st = 128; st > 0; st >>= 1) {
        if ((int)threadIdx.x < st) {
            rs[threadIdx.x] += rs[threadIdx.x + st];
            rq[threadIdx.x] += rq[threadIdx.x + st];
        }
        __syncthreads();
    }
    if (threadIdx.x == 0) {
        float inv = 1.f / (float)rows;
        float m = rs[0] * inv;
        float var = rq[0] * inv - m * m;
        float g  = (j < split) ? gln[j] : gle[j - split];
        float bb = (j < split) ? bln[j] : ble[j - split];
        float s0 = g * rsqrtf(var + 1e-5f);
        scs = s0;
        shs = bb - m * s0;
    }
    __syncthreads();
    for (int r2 = threadIdx.x; r2 < rows; r2 += blockDim.x) {
        size_t idx = (size_t)r2 * cols + j;
        XZ[idx] = XZ[idx] * scs + shs;
    }
}

// Final tiny GEMM: out[s,o] = H[s,:] . c2_w[o,:] + c2_b[o], o in {0,1}
__global__ void c2_kernel(const float* __restrict__ Hin, const float* __restrict__ w,
                          const float* __restrict__ b, float* __restrict__ out, int n)
{
    int gid = blockIdx.x * blockDim.x + threadIdx.x;
    if (gid >= n * 2) return;
    int s = gid >> 1, o = gid & 1;
    const float* h  = Hin + (size_t)s * 1024;
    const float* ww = w + o * 1024;
    float acc = b[o];
    for (int k = 0; k < 1024; ++k) acc = fmaf(h[k], ww[k], acc);
    out[gid] = acc;
}

// ---------------------------------------------------------------------------
// Orchestration
// ---------------------------------------------------------------------------
extern "C" void kernel_launch(void* const* d_in, const int* in_sizes, int n_in,
                              void* d_out, int out_size, void* d_ws, size_t ws_size,
                              hipStream_t stream)
{
    (void)in_sizes; (void)n_in; (void)out_size; (void)ws_size;

    const float* X       = (const float*)d_in[0];
    const float* Z       = (const float*)d_in[1];
    const float* conv1_w = (const float*)d_in[2];
    const float* conv1_b = (const float*)d_in[3];
    const float* conv2_w = (const float*)d_in[4];
    const float* conv2_b = (const float*)d_in[5];
    const float* node_w  = (const float*)d_in[6];
    const float* edge_w  = (const float*)d_in[7];
    const float* bn_n_g  = (const float*)d_in[8];
    const float* bn_n_b  = (const float*)d_in[9];
    const float* bn_e_g  = (const float*)d_in[10];
    const float* bn_e_b  = (const float*)d_in[11];
    const float* ln_w    = (const float*)d_in[12];
    const float* ln_b    = (const float*)d_in[13];
    const float* ln_bn_g = (const float*)d_in[14];
    const float* ln_bn_b = (const float*)d_in[15];
    const float* le_w    = (const float*)d_in[16];
    const float* le_b    = (const float*)d_in[17];
    const float* le_bn_g = (const float*)d_in[18];
    const float* le_bn_b = (const float*)d_in[19];
    const float* c1_w    = (const float*)d_in[20];
    const float* c1_b    = (const float*)d_in[21];
    const float* c2_w    = (const float*)d_in[22];
    const float* c2_b    = (const float*)d_in[23];

    const int n = 2048, R = 116, F = 3, L = 4, H = 128;
    const int DN = R * F;             // 348
    const int DE = R * R;             // 13456
    const int COLS = 4 * H * (L + 1); // 2560

    // workspace layout (f32)
    float* ws = (float*)d_ws;
    size_t off = 0;
    float* Zc  = ws + off; off += (size_t)n * DE;   // mutable Z
    float* Xc  = ws + off; off += (size_t)n * DN;   // mutable X
    float* att = ws + off; off += (size_t)n * DE;   // att; reused as T
    float* V   = ws + off; off += (size_t)n * DE;   // V;   reused as Z2
    float* P   = ws + off; off += (size_t)n * DE;   // P = att @ V
    float* X1  = ws + off; off += (size_t)n * DN;
    float* XZ  = ws + off; off += (size_t)n * COLS;
    float* H1  = ws + off; off += (size_t)n * 1024;
    float* bsc = ws + off; off += R;
    float* bsh = ws + off; off += R;
    float* T   = att;
    float* Z2  = V;

    hipMemcpyAsync(Zc, Z, (size_t)n * DE * sizeof(float), hipMemcpyDeviceToDevice, stream);
    hipMemcpyAsync(Xc, X, (size_t)n * DN * sizeof(float), hipMemcpyDeviceToDevice, stream);

    // batched GEMM grid: one 64x64 tile-block per (sample, 2x2) -> 4 per sample
    const int perSample = ((R + BBM - 1) / BBM) * ((R + BBN - 1) / BBN);  // 4
    const int bgBlocks  = n * perSample;                                  // 8192
    dim3 gB(256);

    // head 0 (X and Z) -> XZ columns [0,128) and [640, 1024)
    {
        dim3 gx((n + GBM - 1) / GBM, (H + GBN - 1) / GBN);
        gemm_bias_relu<<<gx, gB, 0, stream>>>(Xc, DN, ln_w, DN, ln_b,
                                              XZ, COLS, n, H, DN, 1);
        dim3 gz((n + GBM - 1) / GBM, (3 * H + GBN - 1) / GBN);
        gemm_bias_relu<<<gz, gB, 0, stream>>>(Zc, DE, le_w, DE, le_b,
                                              XZ + 5 * H, COLS, n, 3 * H, DE, 1);
    }

    for (int i = 0; i < L; ++i) {
        // attention weights
        attn_kernel<<<n, 128, 0, stream>>>(Xc, conv1_w + i * 9, conv1_b + i * 3, att);

        // V[o,l] = conv2_w[o,:] . Z[l,:] + conv2_b[o]
        bgemm116<<<bgBlocks, 128, 0, stream>>>(conv2_w + (size_t)i * DE, R, 0, 0,
                                               Zc, R, DE, 0,
                                               conv2_b + i * R,
                                               V, R, DE, R, R, R, n);
        // P[l,m] = sum_k att[l,k] V[k,m]
        bgemm116<<<bgBlocks, 128, 0, stream>>>(att, R, DE, 0,
                                               V, R, DE, 1,
                                               nullptr,
                                               P, R, DE, R, R, R, n);
        // T[i,k] = sum_j P[j,i] Z[j,k]   (A-operand = P^T = attention output A)
        bgemm116<<<bgBlocks, 128, 0, stream>>>(P, R, DE, 1,
                                               Zc, R, DE, 1,
                                               nullptr,
                                               T, R, DE, R, R, R, n);
        // Z2 = T @ edge_w
        bgemm116<<<bgBlocks, 128, 0, stream>>>(T, R, DE, 0,
                                               edge_w + (size_t)i * DE, R, 0, 1,
                                               nullptr,
                                               Z2, R, DE, R, R, R, n);
        // Z = relu(bn(Z2)) + Z
        bn_ncl_stats<<<R, 256, 0, stream>>>(Z2, n, R, R, bn_e_g + i * R, bn_e_b + i * R,
                                            bsc, bsh);
        long long totE = (long long)n * DE;
        bn_resid_relu_add<<<(int)((totE + 255) / 256), 256, 0, stream>>>(
            Z2, Zc, bsc, bsh, R, R, totE);

        // edge head -> XZ columns [640 + (i+1)*384, ...)
        {
            dim3 g((n + GBM - 1) / GBM, (3 * H + GBN - 1) / GBN);
            gemm_bias_relu<<<g, gB, 0, stream>>>(Zc, DE,
                                                 le_w + (size_t)(i + 1) * 3 * H * DE, DE,
                                                 le_b + (i + 1) * 3 * H,
                                                 XZ + 5 * H + (i + 1) * 3 * H, COLS,
                                                 n, 3 * H, DE, 1);
        }

        // node branch: X1 = (P^T @ X) @ node_w ; X = relu(bn(X1)) + X
        node_kernel<<<(n * R + 127) / 128, 128, 0, stream>>>(P, Xc, node_w + i * 9, X1, n);
        bn_ncl_stats<<<R, 256, 0, stream>>>(X1, n, R, F, bn_n_g + i * R, bn_n_b + i * R,
                                            bsc, bsh);
        long long totN = (long long)n * DN;
        bn_resid_relu_add<<<(int)((totN + 255) / 256), 256, 0, stream>>>(
            X1, Xc, bsc, bsh, R, F, totN);

        // node head -> XZ columns [(i+1)*128, ...)
        {
            dim3 g((n + GBM - 1) / GBM, (H + GBN - 1) / GBN);
            gemm_bias_relu<<<g, gB, 0, stream>>>(Xc, DN,
                                                 ln_w + (size_t)(i + 1) * H * DN, DN,
                                                 ln_b + (i + 1) * H,
                                                 XZ + (i + 1) * H, COLS,
                                                 n, H, DN, 1);
        }
    }

    // BatchNorm over the concatenated head outputs (commutes with concat)
    bn_cols<<<COLS, 256, 0, stream>>>(XZ, n, COLS, ln_bn_g, ln_bn_b, le_bn_g, le_bn_b, 5 * H);

    // classifier
    {
        dim3 g((n + GBM - 1) / GBM, (1024 + GBN - 1) / GBN);
        gemm_bias_relu<<<g, gB, 0, stream>>>(XZ, COLS, c1_w, COLS, c1_b,
                                             H1, 1024, n, 1024, COLS, 1);
    }
    c2_kernel<<<(n * 2 + 255) / 256, 256, 0, stream>>>(H1, c2_w, c2_b, (float*)d_out, n);
}
